// SEBottleneckInvolution_76020921139657
// MI455X (gfx1250) — compile-verified, matrix operand order checked by reading, not run on hardware
//
#include <hip/hip_runtime.h>
#include <hip/hip_bf16.h>
#include <math.h>

// ---------------------------------------------------------------------------
// SE-Bottleneck with involution, MI455X (gfx1250, wave32, WMMA).
// Activations: pixel-major f16 [HW][Cpad]  -> WMMA A matrix (M=pixels),
//   A-fragment = two unconditional 16B vector loads per lane.
// Weights: pre-packed into WMMA B-fragment order (zero-padded) -> one v16h
//   (2 x b128) load per lane per K-step.
// Epilogue: per-channel affine (BN/bias folded to scale/shift, precomputed;
//   padded channels scale=shift=0) + compile-time activation -> branch-free.
// ---------------------------------------------------------------------------

#define HW   25600        // 160*160
#define Hdim 160
#define Wdim 160
#define EPS  1e-5f
#define KK   11
#define PAD  5

typedef _Float16 half_t;
typedef __attribute__((ext_vector_type(16))) _Float16 v16h;
typedef __attribute__((ext_vector_type(8)))  _Float16 v8h;
typedef __attribute__((ext_vector_type(8)))  float    v8f;

// K-index inside a 16-bit WMMA fragment: element i, lane-half h
//   K = (i&7) + 8*h + 16*(i>>3)
__device__ __forceinline__ int frag_k(int i, int h) {
    return (i & 7) + (h << 3) + ((i >> 3) << 4);
}

__device__ __forceinline__ float elu_f(float x) { return x > 0.f ? x : (expf(x) - 1.f); }

// A fragment: row base pointer (16B aligned), contiguous K runs -> 2 vector loads
__device__ __forceinline__ v16h load_frag_a(const half_t* __restrict__ arow, int k0, int h) {
    const v8h* p0 = (const v8h*)(arow + k0 + (h << 3));
    const v8h* p1 = (const v8h*)(arow + k0 + 16 + (h << 3));
    v8h lo = *p0;
    v8h hi = *p1;
    v16h f;
#pragma unroll
    for (int i = 0; i < 8; ++i) { f[i] = lo[i]; f[i + 8] = hi[i]; }
    return f;
}

// ---------------------------------------------------------------------------
// Weight pre-pack: W (Nout x Kin, row-major f32)  ->  packed B fragments f16.
// Packed index: ((kt*Ntiles + nt)*32 + lane)*16 + i  holds  W^T[k][n]
// with k = kt*32 + frag_k(i, lane>>4), n = nt*16 + (lane&15); OOB -> 0.
// ---------------------------------------------------------------------------
__global__ void pack_wfrag_kernel(const float* __restrict__ W, half_t* __restrict__ P,
                                  int Kin, int Nout, int Ktiles, int Ntiles) {
    int idx = blockIdx.x * blockDim.x + threadIdx.x;
    int total = Ktiles * Ntiles * 512;
    if (idx >= total) return;
    int i    = idx & 15;
    int lane = (idx >> 4) & 31;
    int t    = idx >> 9;
    int nt   = t % Ntiles;
    int kt   = t / Ntiles;
    int h    = lane >> 4;
    int k    = kt * 32 + frag_k(i, h);
    int n    = nt * 16 + (lane & 15);
    float v  = (k < Kin && n < Nout) ? W[n * Kin + k] : 0.f;
    P[idx] = (half_t)v;
}

// ---------------------------------------------------------------------------
// Fold BN (or bias) into per-channel affine scale/shift; padded channels -> 0.
// ---------------------------------------------------------------------------
__global__ void make_scaleshift_kernel(const float* __restrict__ g, const float* __restrict__ b,
                                       const float* __restrict__ m, const float* __restrict__ v,
                                       const float* __restrict__ bias,
                                       float* __restrict__ scale, float* __restrict__ shift,
                                       int Nout, int Npad) {
    int n = blockIdx.x * blockDim.x + threadIdx.x;
    if (n >= Npad) return;
    float sc = 0.f, sh = 0.f;
    if (n < Nout) {
        if (g) { sc = g[n] * rsqrtf(v[n] + EPS); sh = b[n] - m[n] * sc; }
        else   { sc = 1.f; sh = bias ? bias[n] : 0.f; }
    }
    scale[n] = sc;
    shift[n] = sh;
}

// ---------------------------------------------------------------------------
// K1: act0[p*128 + c] = f16( elu( bn1(x[c*HW + p]) ) )   (pixel-major)
// ---------------------------------------------------------------------------
__global__ void bn_elu_to_f16_kernel(const float* __restrict__ x,
                                     const float* __restrict__ g, const float* __restrict__ b,
                                     const float* __restrict__ m, const float* __restrict__ v,
                                     half_t* __restrict__ out, int total) {
    int idx = blockIdx.x * blockDim.x + threadIdx.x;
    if (idx >= total) return;
    int c = idx / HW;
    int p = idx - c * HW;
    float inv = g[c] * rsqrtf(v[c] + EPS);
    float sh  = b[c] - m[c] * inv;
    out[p * 128 + c] = (half_t)elu_f(x[idx] * inv + sh);
}

// ---------------------------------------------------------------------------
// WMMA GEMM: Out = act( (A[HW][Kpad] @ W^T) * scale[n] + shift[n] )
// One wave per 16px x 16ch tile. KTILES = Kpad/32 (fully unrolled).
// ACT: 0=none, 1=ELU, 2=ReLU (compile-time).
// CHANMAJOR=false: Out f16 pixel-major [HW][Npad] (padded cols -> act(0)=0).
// CHANMAJOR=true : Out f32 channel-major [Npad][HW].
// ---------------------------------------------------------------------------
template <typename OutT, int KTILES, bool CHANMAJOR, int ACT>
__global__ void wmma_gemm_kernel(const half_t* __restrict__ A,
                                 const half_t* __restrict__ Bp,   // packed fragments
                                 OutT* __restrict__ Out,
                                 int Kpad, int Npad, int Ntiles,
                                 const float* __restrict__ scale,
                                 const float* __restrict__ shift) {
    int lane = threadIdx.x & 31;
    int wave = threadIdx.x >> 5;
    int nt   = blockIdx.y * (blockDim.x >> 5) + wave;   // out-channel tile
    if (nt >= Ntiles) return;                            // wave-uniform
    int m0   = blockIdx.x * 16;                          // pixel tile
    int h    = lane >> 4;

    const half_t* arow = A + (long)(m0 + (lane & 15)) * Kpad;
    const v16h*   bfr  = (const v16h*)Bp;

    v8f acc = {};
#pragma unroll
    for (int kt = 0; kt < KTILES; ++kt) {
        v16h a = load_frag_a(arow, kt * 32, h);
        v16h b = bfr[((long)kt * Ntiles + nt) * 32 + lane];
        acc = __builtin_amdgcn_wmma_f32_16x16x32_f16(
            false, a, false, b, (short)0, acc, false, false);
    }

    int n = nt * 16 + (lane & 15);   // output channel (invariant over j)
    float sc = scale[n];
    float sh = shift[n];
#pragma unroll
    for (int j = 0; j < 8; ++j) {
        int m = m0 + j + (h << 3);   // pixel
        float o = fmaf(acc[j], sc, sh);
        if (ACT == 1)      o = elu_f(o);
        else if (ACT == 2) o = fmaxf(o, 0.f);
        if (CHANMAJOR) Out[(long)n * HW + m]   = (OutT)o;
        else           Out[(long)m * Npad + n] = (OutT)o;
    }
}

// ---------------------------------------------------------------------------
// Involution + fused bn3 + elu:
//   act3[p*64 + c] = elu(bn3( sum_k wk[p*496 + g*121 + k] * act1[pn*64 + c] ))
// Block = one group x 16-pixel tile; kernel weights staged in LDS.
// Threads: channel-fastest for coalesced pixel-major reads.
// ---------------------------------------------------------------------------
__global__ void involution_kernel(const half_t* __restrict__ act1,
                                  const half_t* __restrict__ wk,
                                  half_t* __restrict__ act3,
                                  const float* __restrict__ g3, const float* __restrict__ b3,
                                  const float* __restrict__ m3, const float* __restrict__ v3) {
    __shared__ half_t wks[16 * 121];
    int tile = blockIdx.x;     // 16 contiguous pixels (within one image row)
    int grp  = blockIdx.y;     // 0..3
    int n0   = tile * 16;

    for (int idx = threadIdx.x; idx < 16 * 121; idx += 256) {
        int px = idx / 121;
        int k  = idx - px * 121;
        wks[idx] = wk[(long)(n0 + px) * 496 + grp * 121 + k];
    }
    __syncthreads();

    int tc = threadIdx.x & 15;    // channel in group (fastest -> coalesced)
    int tx = threadIdx.x >> 4;    // pixel in tile
    int p  = n0 + tx;
    int y  = p / Wdim;
    int x  = p - y * Wdim;
    int c  = grp * 16 + tc;

    float acc = 0.f;
#pragma unroll
    for (int ky = 0; ky < KK; ++ky) {
        int yy = y + ky - PAD;
        if (yy < 0 || yy >= Hdim) continue;
#pragma unroll
        for (int kx = 0; kx < KK; ++kx) {
            int xx = x + kx - PAD;
            if (xx < 0 || xx >= Wdim) continue;
            acc += (float)wks[tx * 121 + ky * KK + kx] *
                   (float)act1[(long)(yy * Wdim + xx) * 64 + c];
        }
    }

    float inv = g3[c] * rsqrtf(v3[c] + EPS);
    float o   = acc * inv + (b3[c] - m3[c] * inv);
    act3[(long)p * 64 + c] = (half_t)elu_f(o);
}

// ---------------------------------------------------------------------------
// Per-channel sum of out3 (channel-major) for SE average pool
// ---------------------------------------------------------------------------
__global__ void colsum_kernel(const float* __restrict__ out3, float* __restrict__ sums) {
    __shared__ float red[256];
    int c = blockIdx.x;
    const float* row = out3 + (long)c * HW;
    float s = 0.f;
    for (int i = threadIdx.x; i < HW; i += 256) s += row[i];
    red[threadIdx.x] = s;
    __syncthreads();
    for (int off = 128; off > 0; off >>= 1) {
        if (threadIdx.x < off) red[threadIdx.x] += red[threadIdx.x + off];
        __syncthreads();
    }
    if (threadIdx.x == 0) sums[c] = red[0];
}

// ---------------------------------------------------------------------------
// SE MLP: 128 -> 8 relu -> 128 sigmoid (one block, 128 threads)
// ---------------------------------------------------------------------------
__global__ void se_kernel(const float* __restrict__ sums,
                          const float* __restrict__ w1, const float* __restrict__ b1,
                          const float* __restrict__ w2, const float* __restrict__ b2,
                          float* __restrict__ s) {
    __shared__ float mean[128];
    __shared__ float h1[8];
    int t = threadIdx.x;
    mean[t] = sums[t] * (1.0f / (float)HW);
    __syncthreads();
    if (t < 8) {
        float a = b1[t];
        for (int c = 0; c < 128; ++c) a += w1[t * 128 + c] * mean[c];
        h1[t] = a > 0.f ? a : 0.f;
    }
    __syncthreads();
    float a = b2[t];
#pragma unroll
    for (int o = 0; o < 8; ++o) a += w2[t * 8 + o] * h1[o];
    s[t] = 1.f / (1.f + expf(-a));
}

// ---------------------------------------------------------------------------
// Final: out = out3 * s[c] + identity(x)   (all channel-major)
// ---------------------------------------------------------------------------
__global__ void final_kernel(const float* __restrict__ out3, const float* __restrict__ s,
                             const float* __restrict__ x, float* __restrict__ out, int total) {
    int idx = blockIdx.x * blockDim.x + threadIdx.x;
    if (idx >= total) return;
    int c = idx / HW;
    out[idx] = out3[idx] * s[c] + x[idx];
}

// ---------------------------------------------------------------------------
// Host side
// ---------------------------------------------------------------------------
extern "C" void kernel_launch(void* const* d_in, const int* in_sizes, int n_in,
                              void* d_out, int out_size, void* d_ws, size_t ws_size,
                              hipStream_t stream) {
    const float* x     = (const float*)d_in[0];
    const float* g1    = (const float*)d_in[1];
    const float* b1    = (const float*)d_in[2];
    const float* m1    = (const float*)d_in[3];
    const float* v1    = (const float*)d_in[4];
    const float* w1    = (const float*)d_in[5];    // (64,128)
    const float* g2    = (const float*)d_in[6];
    const float* b2    = (const float*)d_in[7];
    const float* m2    = (const float*)d_in[8];
    const float* v2    = (const float*)d_in[9];
    const float* invw1 = (const float*)d_in[10];   // (16,64)
    const float* ig    = (const float*)d_in[11];
    const float* ib    = (const float*)d_in[12];
    const float* im    = (const float*)d_in[13];
    const float* iv    = (const float*)d_in[14];
    const float* invw2 = (const float*)d_in[15];   // (484,16)
    const float* invb2 = (const float*)d_in[16];   // (484,)
    const float* g3    = (const float*)d_in[17];
    const float* b3    = (const float*)d_in[18];
    const float* m3    = (const float*)d_in[19];
    const float* v3    = (const float*)d_in[20];
    const float* w3    = (const float*)d_in[21];   // (128,64)
    const float* sew1  = (const float*)d_in[22];   // (8,128)
    const float* seb1  = (const float*)d_in[23];
    const float* sew2  = (const float*)d_in[24];   // (128,8)
    const float* seb2  = (const float*)d_in[25];
    float* out = (float*)d_out;

    // workspace layout (256B aligned offsets)
    char* ws = (char*)d_ws;
    half_t* act0  = (half_t*)(ws + 0);            // [HW][128] f16  6,553,600
    half_t* act1  = (half_t*)(ws + 6553600);      // [HW][64]  f16  3,276,800
    half_t* tbuf  = (half_t*)(ws + 9830400);      // [HW][32]  f16  1,638,400
    half_t* wkb   = (half_t*)(ws + 11468800);     // [HW][496] f16 25,395,200
    half_t* act3  = (half_t*)(ws + 36864000);     // [HW][64]  f16  3,276,800
    float*  out3  = (float*) (ws + 40140800);     // [128][HW] f32 13,107,200
    half_t* pw1   = (half_t*)(ws + 53248000);     // 4*4*512  f16  16,384
    half_t* pinv1 = (half_t*)(ws + 53264384);     // 2*2*512  f16   4,096
    half_t* pinv2 = (half_t*)(ws + 53268480);     // 1*31*512 f16  31,744
    half_t* pw3   = (half_t*)(ws + 53300224);     // 2*8*512  f16  16,384
    float*  sums  = (float*) (ws + 53316608);     // 128 f32
    float*  sbuf  = (float*) (ws + 53317120);     // 128 f32
    float*  sc1   = (float*) (ws + 53317632);     // 64  f32
    float*  sh1   = (float*) (ws + 53317888);     // 64  f32
    float*  sci1  = (float*) (ws + 53318144);     // 32  f32
    float*  shi1  = (float*) (ws + 53318400);     // 32  f32
    float*  sci2  = (float*) (ws + 53318656);     // 496 f32 (2048B slot)
    float*  shi2  = (float*) (ws + 53320704);     // 496 f32 (2048B slot)
    float*  sc3   = (float*) (ws + 53322752);     // 128 f32
    float*  sh3   = (float*) (ws + 53323264);     // 128 f32

    const int TILES = HW / 16;  // 1600 pixel tiles

    // P0: pack weights into WMMA B-fragment order (tiny)
    pack_wfrag_kernel<<<(4 * 4 * 512 + 255) / 256, 256, 0, stream>>>(w1,    pw1,   128, 64,  4, 4);
    pack_wfrag_kernel<<<(2 * 2 * 512 + 255) / 256, 256, 0, stream>>>(invw1, pinv1, 64,  16,  2, 2);
    pack_wfrag_kernel<<<(1 * 31 * 512 + 255) / 256, 256, 0, stream>>>(invw2, pinv2, 16, 484, 1, 31);
    pack_wfrag_kernel<<<(2 * 8 * 512 + 255) / 256, 256, 0, stream>>>(w3,    pw3,   64, 128,  2, 8);

    // P1: fold BN / bias into per-channel affine (padded channels -> 0)
    make_scaleshift_kernel<<<1, 64,  0, stream>>>(g2, b2, m2, v2, nullptr, sc1,  sh1,  64, 64);
    make_scaleshift_kernel<<<1, 32,  0, stream>>>(ig, ib, im, iv, nullptr, sci1, shi1, 16, 32);
    make_scaleshift_kernel<<<2, 256, 0, stream>>>(nullptr, nullptr, nullptr, nullptr, invb2,
                                                  sci2, shi2, 484, 496);
    make_scaleshift_kernel<<<1, 128, 0, stream>>>(nullptr, nullptr, nullptr, nullptr, nullptr,
                                                  sc3, sh3, 128, 128);

    // K1: act0 = f16(elu(bn1(x))), pixel-major
    {
        int total = 128 * HW;
        bn_elu_to_f16_kernel<<<(total + 255) / 256, 256, 0, stream>>>(
            x, g1, b1, m1, v1, act0, total);
    }
    // K2: act1 = f16(elu(bn2(act0 @ w1^T)))   K=128, N=64
    wmma_gemm_kernel<half_t, 4, false, 1><<<dim3(TILES, 1), 128, 0, stream>>>(
        act0, pw1, act1, 128, 64, 4, sc1, sh1);
    // K3: t = f16(relu(bn_i(act1 @ inv_w1^T)))  K=64, N=16 (stored padded to 32)
    wmma_gemm_kernel<half_t, 2, false, 2><<<dim3(TILES, 1), 64, 0, stream>>>(
        act1, pinv1, tbuf, 64, 32, 2, sci1, shi1);
    // K4: wk = f16(t @ inv_w2^T + inv_b2)       K=32(zero-padded), N=484 (pad 496)
    wmma_gemm_kernel<half_t, 1, false, 0><<<dim3(TILES, 8), 128, 0, stream>>>(
        tbuf, pinv2, wkb, 32, 496, 31, sci2, shi2);
    // K5: involution + bn3 + elu -> act3
    involution_kernel<<<dim3(TILES, 4), 256, 0, stream>>>(
        act1, wkb, act3, g3, b3, m3, v3);
    // K6: out3 = act3 @ w3^T (f32, channel-major)  K=64, N=128
    wmma_gemm_kernel<float, 2, true, 0><<<dim3(TILES, 2), 128, 0, stream>>>(
        act3, pw3, out3, 64, 128, 8, sc3, sh3);
    // K7: channel sums for SE
    colsum_kernel<<<128, 256, 0, stream>>>(out3, sums);
    // K8: SE MLP -> per-channel sigmoid scale
    se_kernel<<<1, 128, 0, stream>>>(sums, sew1, seb1, sew2, seb2, sbuf);
    // K9: out = out3 * s + x
    {
        int total = 128 * HW;
        final_kernel<<<(total + 255) / 256, 256, 0, stream>>>(out3, sbuf, x, out, total);
    }
    (void)in_sizes; (void)n_in; (void)out_size; (void)ws_size;
}
